// DecoderLSTM_7928509628518
// MI455X (gfx1250) — matrix operand order repaired and checked by reading, not compiled, not run on hardware
//
#include <hip/hip_runtime.h>
#include <hip/hip_bf16.h>

// ---------------------------------------------------------------------------
// DecoderLSTM for MI455X (gfx1250, wave32, WMMA).
//
// Dominant cost: preds = h_t @ lin_W.T  (M=3264, N=32000, K=512, 107 GFLOP,
// 418 MB fp32 output).  Strategy:
//   * lin_W / W_ih / W_hh converted once to bf16 in workspace; bf16 lin_W is
//     32.8 MB -> resident in the 192 MB L2 across all 204 M-tile passes.
//   * v_wmma_f32_16x16x32_bf16 for all GEMMs (fp32 16x16x4 WMMA would be
//     ~8x lower MAC throughput and push us compute-bound; bf16 keeps the
//     kernel store-bandwidth-bound at ~18us @ 23.3 TB/s).
//   * non-temporal stores for the 418 MB prediction tensor so it does not
//     evict lin_W from L2.
//   * LSTM recurrence: 51 sequential (gates-GEMM + pointwise) launches.
// ---------------------------------------------------------------------------

typedef __attribute__((ext_vector_type(16))) __bf16          bf16x16;
typedef __attribute__((ext_vector_type(8)))  float           v8f;
typedef __attribute__((ext_vector_type(8)))  unsigned short  u16x8;
typedef __attribute__((ext_vector_type(16))) unsigned short  u16x16;

#define DEV static __device__ __forceinline__

// float -> bf16, round-to-nearest-even
DEV unsigned short f2bf(float f) {
  unsigned u = __builtin_bit_cast(unsigned, f);
  u += 0x7FFFu + ((u >> 16) & 1u);
  return (unsigned short)(u >> 16);
}

DEV float sigm(float x) { return 1.0f / (1.0f + expf(-x)); }

// A-matrix fragment (16x32 bf16, k order shared with load_b_frag): lane holds row M = lane%16.
// element e: K = 16*(lane/16) + e  -> two contiguous 16B loads.
DEV bf16x16 load_a_frag(const unsigned short* __restrict__ base,
                        int row, int ld, int kbase, int hi) {
  const unsigned short* p = base + (size_t)row * ld + kbase + 16 * hi;
  u16x8 lo = *(const u16x8*)(p);
  u16x8 up = *(const u16x8*)(p + 8);
  u16x16 v;
#pragma unroll
  for (int e = 0; e < 8; ++e) { v[e] = lo[e]; v[e + 8] = up[e]; }
  return __builtin_bit_cast(bf16x16, v);
}

// B-matrix fragment (32x16 bf16): lane holds column N = lane%16 (a row of the
// row-major weight matrix W[n,k]); element e: K = 16*(lane/16) + e
// -> one contiguous 32B load.
DEV bf16x16 load_b_frag(const unsigned short* __restrict__ base,
                        int row, int ld, int kbase, int hi) {
  u16x16 v = *(const u16x16*)(base + (size_t)row * ld + kbase + 16 * hi);
  return __builtin_bit_cast(bf16x16, v);
}

DEV v8f wmma_bf16(bf16x16 a, bf16x16 b, v8f c) {
  // (neg_a, A, neg_b, B, c_mod, C, reuse_a, reuse_b)
  return __builtin_amdgcn_wmma_f32_16x16x32_bf16(false, a, false, b,
                                                 (short)0, c, false, false);
}

// ---------------------------------------------------------------------------
// Setup kernels
// ---------------------------------------------------------------------------

__global__ void k_f32_to_bf16(const float* __restrict__ s,
                              unsigned short* __restrict__ d, int n) {
  int i = blockIdx.x * blockDim.x + threadIdx.x;
  if (i < n) d[i] = f2bf(s[i]);
}

// Stable descending sort of 64 lengths (matches jnp.argsort(-len), stable).
__global__ void k_sort(const int* __restrict__ cap_len,
                       int* __restrict__ sortind, int* __restrict__ declen) {
  __shared__ int len[64];
  int i = threadIdx.x;
  len[i] = cap_len[i];
  __syncthreads();
  int li = len[i];
  int rank = 0;
#pragma unroll 8
  for (int j = 0; j < 64; ++j) {
    int lj = len[j];
    rank += (lj > li) || (lj == li && j < i);
  }
  sortind[rank] = i;
  declen[rank]  = li - 1;
}

__global__ void k_gather_init(const float* __restrict__ img,
                              const int* __restrict__ caps,
                              const int* __restrict__ sortind,
                              const int* __restrict__ declen,
                              int* __restrict__ caps_sorted,
                              float* __restrict__ Hf,
                              unsigned short* __restrict__ Hbf,
                              float* __restrict__ Cst,
                              float* __restrict__ out_caps,
                              float* __restrict__ out_declen,
                              float* __restrict__ out_sortind) {
  int k = blockIdx.x;           // sorted position
  int src = sortind[k];
  for (int t = threadIdx.x; t < 52; t += blockDim.x) {
    int c = caps[src * 52 + t];
    caps_sorted[k * 52 + t] = c;
    out_caps[k * 52 + t] = (float)c;
  }
  if (threadIdx.x == 0) {
    out_declen[k]  = (float)declen[k];
    out_sortind[k] = (float)src;
  }
  for (int e = threadIdx.x; e < 512; e += blockDim.x) {
    float h = img[src * 512 + e];   // h0 = sorted image features
    Hf[k * 512 + e]  = h;
    Hbf[k * 512 + e] = f2bf(h);
    Cst[k * 512 + e] = 0.0f;        // c0 = 0
  }
}

// X[t][b][k] = bf16(emb_W[caps_sorted[b,t]][k]), t-major for the step GEMM.
__global__ void k_embed(const float* __restrict__ embW,
                        const int* __restrict__ caps_sorted,
                        unsigned short* __restrict__ X) {
  int bt = blockIdx.x;            // b*51 + t
  int b = bt / 51, t = bt - b * 51;
  int cap = caps_sorted[b * 52 + t];
  const float* src = embW + (size_t)cap * 512;
  unsigned short* dst = X + ((size_t)t * 64 + b) * 512;
  for (int e = threadIdx.x; e < 512; e += blockDim.x) dst[e] = f2bf(src[e]);
}

// ---------------------------------------------------------------------------
// gates = x_t @ W_ih.T + h_{t-1} @ W_hh.T   (M=64, N=2048, K=512+512)
// 512 16x16 tiles, 4 waves/block, one tile per wave, 32 WMMAs each.
// ---------------------------------------------------------------------------
__global__ void __launch_bounds__(128)
k_gates_gemm(const unsigned short* __restrict__ Xt,
             const unsigned short* __restrict__ Hc,
             const unsigned short* __restrict__ Wih,
             const unsigned short* __restrict__ Whh,
             float* __restrict__ gates) {
  int wave = threadIdx.x >> 5, lane = threadIdx.x & 31;
  int job = blockIdx.x * 4 + wave;            // 0..511
  int mt = job & 3, nt = job >> 2;            // 4 M-tiles x 128 N-tiles
  int hi = lane >> 4, lm = lane & 15;
  int arow = mt * 16 + lm;                    // batch row
  int brow = nt * 16 + lm;                    // gate output n
  v8f c = {};
#pragma unroll 4
  for (int kk = 0; kk < 16; ++kk) {
    bf16x16 a = load_a_frag(Xt, arow, 512, kk * 32, hi);
    bf16x16 b = load_b_frag(Wih, brow, 512, kk * 32, hi);
    c = wmma_bf16(a, b, c);
  }
#pragma unroll 4
  for (int kk = 0; kk < 16; ++kk) {
    bf16x16 a = load_a_frag(Hc, arow, 512, kk * 32, hi);
    bf16x16 b = load_b_frag(Whh, brow, 512, kk * 32, hi);
    c = wmma_bf16(a, b, c);
  }
#pragma unroll
  for (int v = 0; v < 8; ++v) {               // C: (M=v+8*hi, N=lm)
    int m = mt * 16 + v + 8 * hi;
    gates[m * 2048 + nt * 16 + lm] = c[v];
  }
}

// ---------------------------------------------------------------------------
// LSTM pointwise: i,f,g,o -> c,h with masking; emit bf16 h for next GEMM and
// for the classifier (H_all row r = b*51 + t).
// ---------------------------------------------------------------------------
__global__ void k_lstm_point(const float* __restrict__ gates,
                             const float* __restrict__ b_ih,
                             const float* __restrict__ b_hh,
                             float* __restrict__ Cst,
                             float* __restrict__ Hf,
                             unsigned short* __restrict__ Hbf,
                             unsigned short* __restrict__ Hall,
                             const int* __restrict__ declen, int t) {
  int idx = blockIdx.x * blockDim.x + threadIdx.x;   // 0..32767
  int b = idx >> 9, k = idx & 511;
  const float* gb = gates + b * 2048;
  float gi = gb[k]        + b_ih[k]        + b_hh[k];
  float gf = gb[512 + k]  + b_ih[512 + k]  + b_hh[512 + k];
  float gg = gb[1024 + k] + b_ih[1024 + k] + b_hh[1024 + k];
  float go = gb[1536 + k] + b_ih[1536 + k] + b_hh[1536 + k];
  float i = sigm(gi), f = sigm(gf), g = tanhf(gg), o = sigm(go);
  float cn = f * Cst[idx] + i * g;
  float hn = o * tanhf(cn);
  bool act = t < declen[b];
  float hc = act ? hn : Hf[idx];
  float cc = act ? cn : Cst[idx];
  Cst[idx] = cc;
  Hf[idx]  = hc;
  Hbf[idx] = f2bf(hc);                                  // carried h (next step)
  Hall[((size_t)b * 51 + t) * 512 + k] = f2bf(hn);      // masked later
}

// ---------------------------------------------------------------------------
// preds = H_all @ lin_W.T + lin_b, masked.  M=3264, N=32000, K=512.
// Each wave: one 16M x 64N tile (A fragment reused across 4 WMMAs/K-step).
// 204 M-tiles x 500 N-groups = 102000 wave jobs, 4 waves/block.
// ---------------------------------------------------------------------------
__global__ void __launch_bounds__(128)
k_preds_gemm(const unsigned short* __restrict__ Hall,
             const unsigned short* __restrict__ Wl,
             const float* __restrict__ lin_b,
             const int* __restrict__ declen,
             float* __restrict__ out) {
  int wave = threadIdx.x >> 5, lane = threadIdx.x & 31;
  int job = blockIdx.x * 4 + wave;            // 0..101999
  int mt = job / 500, ng = job - mt * 500;    // block-adjacent jobs share A
  int hi = lane >> 4, lm = lane & 15;
  int arow = mt * 16 + lm;
  int nb = ng * 64;
  v8f acc[4] = {};
#pragma unroll 2
  for (int kk = 0; kk < 16; ++kk) {
    bf16x16 a  = load_a_frag(Hall, arow, 512, kk * 32, hi);
    bf16x16 b0 = load_b_frag(Wl, nb + 0  + lm, 512, kk * 32, hi);
    bf16x16 b1 = load_b_frag(Wl, nb + 16 + lm, 512, kk * 32, hi);
    bf16x16 b2 = load_b_frag(Wl, nb + 32 + lm, 512, kk * 32, hi);
    bf16x16 b3 = load_b_frag(Wl, nb + 48 + lm, 512, kk * 32, hi);
    acc[0] = wmma_bf16(a, b0, acc[0]);
    acc[1] = wmma_bf16(a, b1, acc[1]);
    acc[2] = wmma_bf16(a, b2, acc[2]);
    acc[3] = wmma_bf16(a, b3, acc[3]);
  }
#pragma unroll
  for (int v = 0; v < 8; ++v) {
    int r = mt * 16 + v + 8 * hi;             // row = b*51 + t
    int bb = r / 51, t = r - bb * 51;
    bool act = t < declen[bb];
    size_t rowoff = (size_t)r * 32000;
#pragma unroll
    for (int j = 0; j < 4; ++j) {
      int col = nb + j * 16 + lm;
      float val = act ? (acc[j][v] + lin_b[col]) : 0.0f;
      // NT store: don't let the 418 MB output evict bf16 lin_W from L2
      __builtin_nontemporal_store(val, out + rowoff + col);
    }
  }
}

// ---------------------------------------------------------------------------
// Host launcher
// ---------------------------------------------------------------------------
extern "C" void kernel_launch(void* const* d_in, const int* in_sizes, int n_in,
                              void* d_out, int out_size, void* d_ws,
                              size_t ws_size, hipStream_t stream) {
  (void)in_sizes; (void)n_in; (void)out_size;

  const float* img   = (const float*)d_in[0];   // (64, 512)
  const int*   caps  = (const int*)  d_in[1];   // (64, 52)
  const int*   clen  = (const int*)  d_in[2];   // (64, 1)
  const float* embW  = (const float*)d_in[3];   // (32000, 512)
  const float* W_ih  = (const float*)d_in[4];   // (2048, 512)
  const float* W_hh  = (const float*)d_in[5];   // (2048, 512)
  const float* b_ih  = (const float*)d_in[6];   // (2048,)
  const float* b_hh  = (const float*)d_in[7];   // (2048,)
  const float* lin_W = (const float*)d_in[8];   // (32000, 512)
  const float* lin_b = (const float*)d_in[9];   // (32000,)
  float* out = (float*)d_out;

  // workspace layout (256B aligned slabs)
  char* ws = (char*)d_ws;
  size_t off = 0;
  auto take = [&](size_t bytes) -> char* {
    char* p = ws + off;
    off = (off + bytes + 255) & ~(size_t)255;
    return p;
  };
  unsigned short* linw_bf = (unsigned short*)take(32000ull * 512 * 2);
  unsigned short* wih_bf  = (unsigned short*)take(2048ull * 512 * 2);
  unsigned short* whh_bf  = (unsigned short*)take(2048ull * 512 * 2);
  unsigned short* X       = (unsigned short*)take(51ull * 64 * 512 * 2);
  unsigned short* Hall    = (unsigned short*)take(3264ull * 512 * 2);
  unsigned short* Hbf     = (unsigned short*)take(64ull * 512 * 2);
  float* Hf    = (float*)take(64ull * 512 * 4);
  float* Cst   = (float*)take(64ull * 512 * 4);
  float* gates = (float*)take(64ull * 2048 * 4);
  int* sortind     = (int*)take(64 * 4);
  int* declen      = (int*)take(64 * 4);
  int* caps_sorted = (int*)take(64ull * 52 * 4);
  if (off > ws_size) return;  // ~45 MB needed

  // output regions (tuple concatenated flat, ints emitted as float values)
  const size_t preds_n   = 64ull * 51 * 32000;   // 104,448,000
  float* out_caps    = out + preds_n;            // (64, 52)
  float* out_declen  = out_caps + 64 * 52;       // (64,)
  float* out_sortind = out_declen + 64;          // (64,)

  // 1) weights -> bf16 (once per launch; deterministic)
  k_f32_to_bf16<<<(32000 * 512 + 255) / 256, 256, 0, stream>>>(
      lin_W, linw_bf, 32000 * 512);
  k_f32_to_bf16<<<(2048 * 512 + 255) / 256, 256, 0, stream>>>(
      W_ih, wih_bf, 2048 * 512);
  k_f32_to_bf16<<<(2048 * 512 + 255) / 256, 256, 0, stream>>>(
      W_hh, whh_bf, 2048 * 512);

  // 2) stable descending sort, gather, h0/c0 init, embedding
  k_sort<<<1, 64, 0, stream>>>(clen, sortind, declen);
  k_gather_init<<<64, 256, 0, stream>>>(img, caps, sortind, declen,
                                        caps_sorted, Hf, Hbf, Cst,
                                        out_caps, out_declen, out_sortind);
  k_embed<<<64 * 51, 256, 0, stream>>>(embW, caps_sorted, X);

  // 3) sequential LSTM recurrence (51 steps)
  for (int t = 0; t < 51; ++t) {
    k_gates_gemm<<<128, 128, 0, stream>>>(X + (size_t)t * 64 * 512, Hbf,
                                          wih_bf, whh_bf, gates);
    k_lstm_point<<<128, 256, 0, stream>>>(gates, b_ih, b_hh, Cst, Hf, Hbf,
                                          Hall, declen, t);
  }

  // 4) classifier GEMM over all (b, t) at once — the 107 GFLOP bulk
  k_preds_gemm<<<25500, 128, 0, stream>>>(Hall, linw_bf, lin_b, declen, out);
}
